// ContinuousThoughtMachine_34213709480142
// MI455X (gfx1250) — matrix operand order repaired and checked by reading, not compiled, not run on hardware
//
#include <hip/hip_runtime.h>
#include <hip/hip_bf16.h>
#include <math.h>

typedef _Float16 h16;
typedef __attribute__((ext_vector_type(16))) _Float16 v16h;
typedef __attribute__((ext_vector_type(8)))  _Float16 v8h;
typedef __attribute__((ext_vector_type(8)))  float    v8f;

#define BATCH   64
#define DMODEL  1024
#define HDIM    32
#define NITERS  16
#define SYNCH   32896
#define OUTD    1000
#define OUTP    1024
#define EPSV    1e-5f

// ---------------------------------------------------------------------------
// WMMA GEMM: C[M,N] += A[M,K](f16 row-major) x Bt[N,K](f16, K-contiguous)
// split-K partials (deterministic), f32 accumulate.  M%16==0, N%256==0, K%32==0.
// block = 128 threads (4 waves); wave w owns a 16x64 strip (4 WMMA tiles) so
// the A fragment is loaded once per k-step and feeds 4 independent WMMAs.
// Ping-pong software pipeline (unroll-2, disjoint register sets): no register
// rotation movs, so no WMMA->VALU hazard NOPs in the steady-state loop.
// ---------------------------------------------------------------------------
__global__ void gemm_wmma_splitk(const h16* __restrict__ A, const h16* __restrict__ Bt,
                                 float* __restrict__ Cpart,
                                 int M, int N, int K, int kSteps, int stepsPerPart)
{
    const int lane = threadIdx.x & 31;
    const int wave = threadIdx.x >> 5;
    const int m0   = blockIdx.y << 4;
    const int n0   = (blockIdx.x << 8) + (wave << 6);   // 64-wide strip per wave
    const int part = blockIdx.z;
    int ks0 = part * stepsPerPart;
    int ks1 = ks0 + stepsPerPart; if (ks1 > kSteps) ks1 = kSteps;
    const int nIt = ks1 - ks0;

    v8f c0 = {0.f,0.f,0.f,0.f,0.f,0.f,0.f,0.f};
    v8f c1 = c0, c2 = c0, c3 = c0;
    const int l15 = lane & 15;
    const int hi  = lane >> 4;                  // 0 for lanes 0-15, 1 for 16-31
    const h16* aP  = A  + (size_t)(m0 + l15) * K + (hi << 4);   // +0 or +16
    const h16* bP0 = Bt + (size_t)(n0 + l15) * K + (hi << 4);   // +0 or +16
    const h16* bP1 = bP0 + (size_t)16 * K;
    const h16* bP2 = bP0 + (size_t)32 * K;
    const h16* bP3 = bP0 + (size_t)48 * K;

    union AF { v16h v; v8h h[2]; };
    AF a0, a1;
    v16h b00, b01, b02, b03;
    v16h b10, b11, b12, b13;

#define LOADF(AA, B0, B1, B2, B3, KS)                                  \
    {                                                                  \
        const int k0_ = (KS) << 5;                                     \
        AA.h[0] = *(const v8h*)(aP + k0_);                             \
        AA.h[1] = *(const v8h*)(aP + k0_ + 8);                         \
        B0 = *(const v16h*)(bP0 + k0_);                                \
        B1 = *(const v16h*)(bP1 + k0_);                                \
        B2 = *(const v16h*)(bP2 + k0_);                                \
        B3 = *(const v16h*)(bP3 + k0_);                                \
    }
#define WMMA4(AA, B0, B1, B2, B3)                                      \
    c0 = __builtin_amdgcn_wmma_f32_16x16x32_f16(false, AA.v, false, B0, (short)0, c0, false, false); \
    c1 = __builtin_amdgcn_wmma_f32_16x16x32_f16(false, AA.v, false, B1, (short)0, c1, false, false); \
    c2 = __builtin_amdgcn_wmma_f32_16x16x32_f16(false, AA.v, false, B2, (short)0, c2, false, false); \
    c3 = __builtin_amdgcn_wmma_f32_16x16x32_f16(false, AA.v, false, B3, (short)0, c3, false, false);

    if (nIt > 0) {
        LOADF(a0, b00, b01, b02, b03, ks0);
        int i = 0;
        for (; i + 2 <= nIt; i += 2) {
            LOADF(a1, b10, b11, b12, b13, ks0 + i + 1);
            WMMA4(a0, b00, b01, b02, b03);
            if (i + 2 < nIt) LOADF(a0, b00, b01, b02, b03, ks0 + i + 2);
            WMMA4(a1, b10, b11, b12, b13);
        }
        if (i < nIt) {                       // odd tail: data already in buf0
            WMMA4(a0, b00, b01, b02, b03);
        }
    }
#undef LOADF
#undef WMMA4

    float* cp = Cpart + (size_t)part * M * N
                      + (size_t)(m0 + (hi << 3)) * N + (n0 + l15);
#pragma unroll
    for (int v = 0; v < 8; ++v) {
        cp[(size_t)v * N + 0 ] = c0[v];
        cp[(size_t)v * N + 16] = c1[v];
        cp[(size_t)v * N + 32] = c2[v];
        cp[(size_t)v * N + 48] = c3[v];
    }
}

__global__ void reduce_bias_k(const float* __restrict__ Cpart, int parts, size_t MN,
                              const float* __restrict__ bias, int biasN,
                              float* __restrict__ outF, h16* __restrict__ outH, int N)
{
    size_t idx = (size_t)blockIdx.x * 256 + threadIdx.x;
    if (idx >= MN) return;
    float s = 0.f;
    for (int p = 0; p < parts; ++p) s += Cpart[(size_t)p * MN + idx];
    int n = (int)(idx % (size_t)N);
    if (bias && n < biasN) s += bias[n];
    if (outF) outF[idx] = s;
    if (outH) outH[idx] = (h16)s;
}

// W[K,N] f32 -> Wt[Npad,K] f16 (zero pad columns n>=N)
__global__ void w_transpose_h(const float* __restrict__ W, h16* __restrict__ Wt,
                              int K, int N, int Npad)
{
    size_t idx = (size_t)blockIdx.x * 256 + threadIdx.x;
    size_t total = (size_t)Npad * K;
    if (idx >= total) return;
    int k = (int)(idx % (size_t)K);
    int n = (int)(idx / (size_t)K);
    float v = (n < N) ? W[(size_t)k * N + n] : 0.f;
    Wt[idx] = (h16)v;
}

// ---------------------------------------------------------------------------
// conv stem
// ---------------------------------------------------------------------------
__global__ void conv3x3_k(const float* __restrict__ in, const float* __restrict__ w,
                          const float* __restrict__ bias, float* __restrict__ out,
                          int Cin, int HW)
{
    int b  = blockIdx.z;
    int oc = blockIdx.y;
    int pix = blockIdx.x * 256 + threadIdx.x;
    if (pix >= HW * HW) return;
    int oy = pix / HW, ox = pix - oy * HW;
    const float* wb = w + (size_t)oc * Cin * 9;
    float acc = bias[oc];
    for (int ic = 0; ic < Cin; ++ic) {
        const float* ip = in + ((size_t)(b * Cin + ic)) * HW * HW;
        const float* wp = wb + ic * 9;
#pragma unroll
        for (int ky = 0; ky < 3; ++ky) {
            int iy = oy + ky - 1;
            if (iy < 0 || iy >= HW) continue;
#pragma unroll
            for (int kx = 0; kx < 3; ++kx) {
                int ix = ox + kx - 1;
                if (ix < 0 || ix >= HW) continue;
                acc += ip[iy * HW + ix] * wp[ky * 3 + kx];
            }
        }
    }
    out[((size_t)(b * 256 + oc)) * HW * HW + pix] = acc;
}

__global__ void bn_stats_k(const float* __restrict__ x, float* __restrict__ stats, int HW2)
{
    int c = blockIdx.x;
    int tot = BATCH * HW2;
    float s = 0.f, q = 0.f;
    for (int i = threadIdx.x; i < tot; i += 256) {
        int b = i / HW2, p = i - b * HW2;
        float v = x[((size_t)(b * 256 + c)) * HW2 + p];
        s += v; q += v * v;
    }
    __shared__ float rs[256], rq[256];
    rs[threadIdx.x] = s; rq[threadIdx.x] = q; __syncthreads();
    for (int off = 128; off > 0; off >>= 1) {
        if ((int)threadIdx.x < off) { rs[threadIdx.x] += rs[threadIdx.x + off]; rq[threadIdx.x] += rq[threadIdx.x + off]; }
        __syncthreads();
    }
    if (threadIdx.x == 0) {
        float inv = 1.f / (float)tot;
        float m = rs[0] * inv;
        float var = rq[0] * inv - m * m;
        stats[c] = m; stats[256 + c] = rsqrtf(var + EPSV);
    }
}

__global__ void bn_relu_pool_k(const float* __restrict__ y, const float* __restrict__ stats,
                               const float* __restrict__ g, const float* __restrict__ bb,
                               float* __restrict__ out, int HW)
{
    int OH = HW >> 1;
    size_t idx = (size_t)blockIdx.x * 256 + threadIdx.x;
    size_t total = (size_t)BATCH * 256 * OH * OH;
    if (idx >= total) return;
    int ox = (int)(idx % OH); size_t t = idx / OH;
    int oy = (int)(t % OH); t /= OH;
    int c  = (int)(t % 256); int b = (int)(t / 256);
    float m = stats[c], rstd = stats[256 + c], gg = g[c], be = bb[c];
    const float* yp = y + ((size_t)(b * 256 + c)) * HW * HW;
    float best = 0.f;
#pragma unroll
    for (int dy = 0; dy < 2; ++dy)
#pragma unroll
        for (int dx = 0; dx < 2; ++dx) {
            float v = yp[(2 * oy + dy) * HW + 2 * ox + dx];
            v = gg * (v - m) * rstd + be;
            v = v > 0.f ? v : 0.f;
            best = v > best ? v : best;
        }
    out[idx] = best;
}

// feat16[(b*256+s)*256+d] = h2[(b*256+d)*256+s]
__global__ void feat16_k(const float* __restrict__ h2, h16* __restrict__ feat)
{
    size_t idx = (size_t)blockIdx.x * 256 + threadIdx.x;  // b,s,d
    int d = (int)(idx & 255);
    int s = (int)((idx >> 8) & 255);
    int b = (int)(idx >> 16);
    feat[idx] = (h16)h2[((size_t)(b * 256 + d)) * 256 + s];
}

// LayerNorm over 256-wide rows, optional f32 + f16 outputs (in-place safe)
__global__ void ln256_k(const float* __restrict__ in, const float* __restrict__ g,
                        const float* __restrict__ b, float* __restrict__ outF,
                        h16* __restrict__ outH)
{
    int row = blockIdx.x, c = threadIdx.x;
    float v = in[(size_t)row * 256 + c];
    __shared__ float red[256];
    red[c] = v; __syncthreads();
    for (int off = 128; off > 0; off >>= 1) { if (c < off) red[c] += red[c + off]; __syncthreads(); }
    float m = red[0] * (1.f / 256.f); __syncthreads();
    float d = v - m;
    red[c] = d * d; __syncthreads();
    for (int off = 128; off > 0; off >>= 1) { if (c < off) red[c] += red[c + off]; __syncthreads(); }
    float rstd = rsqrtf(red[0] * (1.f / 256.f) + EPSV);
    float o = g[c] * d * rstd + b[c];
    if (outF) outF[(size_t)row * 256 + c] = o;
    if (outH) outH[(size_t)row * 256 + c] = (h16)o;
}

// ---------------------------------------------------------------------------
// synchrony machinery
// ---------------------------------------------------------------------------
__global__ void triu_init_k(int* __restrict__ ti, int* __restrict__ tj)
{
    int p = blockIdx.x * 256 + threadIdx.x;
    if (p >= SYNCH) return;
    float disc = 263169.f - 8.f * (float)p;           // (2N+1)^2 - 8p, N=256
    int i = (int)((513.f - sqrtf(disc)) * 0.5f);
    if (i > 255) i = 255; if (i < 0) i = 0;
    while (i > 0 && (i * (513 - i)) / 2 > p) --i;
    while (i < 255 && ((i + 1) * (513 - (i + 1))) / 2 <= p) ++i;
    int j = i + (p - (i * (513 - i)) / 2);
    ti[p] = i; tj[p] = j;
}

__global__ void exp_neg_k(const float* __restrict__ d, float* __restrict__ r)
{
    int p = blockIdx.x * 256 + threadIdx.x;
    if (p < SYNCH) r[p] = expf(-d[p]);
}

__global__ void init_state_k(const float* __restrict__ start_act,
                             const float* __restrict__ start_trace,
                             float* __restrict__ act, float* __restrict__ trace)
{
    int idx = blockIdx.x * 256 + threadIdx.x;            // b*1024+d
    int d = idx & 1023;
    act[idx] = start_act[d];
    float* tr = trace + (size_t)idx * 25;
    const float* st = start_trace + (size_t)d * 25;
    for (int m = 0; m < 25; ++m) tr[m] = st[m];
}

__global__ void init_syn_k(const float* __restrict__ start_act,
                           const int* __restrict__ ti, const int* __restrict__ tj,
                           float* __restrict__ aa, float* __restrict__ ba,
                           float* __restrict__ ao, float* __restrict__ bo)
{
    int p = blockIdx.x * 256 + threadIdx.x;
    if (p >= SYNCH) return;
    int b = blockIdx.y;
    size_t id = (size_t)b * SYNCH + p;
    aa[id] = 0.f; ba[id] = 0.f;
    ao[id] = start_act[ti[p]] * start_act[tj[p]];        // sel0 = act0[:, :256]
    bo[id] = 1.f;
}

__global__ void synch_k(const float* __restrict__ act, int selOff,
                        const int* __restrict__ ti, const int* __restrict__ tj,
                        const float* __restrict__ r,
                        float* __restrict__ alpha, float* __restrict__ beta,
                        float* __restrict__ sF, h16* __restrict__ sH)
{
    int p = blockIdx.x * 256 + threadIdx.x;
    if (p >= SYNCH) return;
    int b = blockIdx.y;
    const float* a = act + (size_t)b * DMODEL + selOff;
    float pp = a[ti[p]] * a[tj[p]];
    size_t id = (size_t)b * SYNCH + p;
    float rr = r[p];
    float al = rr * alpha[id] + pp;
    float be = rr * beta[id] + 1.f;
    alpha[id] = al; beta[id] = be;
    float s = al * rsqrtf(be);
    if (sF) sF[id] = s;
    sH[id] = (h16)s;
}

// ---------------------------------------------------------------------------
// attention (B x 8 heads, 256 tokens, head_dim 32)
// ---------------------------------------------------------------------------
__global__ void attention_k(const float* __restrict__ qh, const float* __restrict__ kh,
                            const float* __restrict__ vh, float* __restrict__ attn)
{
    int b = blockIdx.x, h = blockIdx.y, s = threadIdx.x;
    __shared__ float sc[256], red[256];
    const float* q = qh + (size_t)b * 256 + h * HDIM;
    const float* k = kh + ((size_t)(b * 256 + s)) * 256 + h * HDIM;
    float d = 0.f;
#pragma unroll
    for (int e = 0; e < HDIM; ++e) d += q[e] * k[e];
    d *= 0.17677669529663687f;                       // 1/sqrt(32)
    sc[s] = d; red[s] = d; __syncthreads();
    for (int off = 128; off > 0; off >>= 1) { if (s < off) red[s] = fmaxf(red[s], red[s + off]); __syncthreads(); }
    float mx = red[0]; __syncthreads();
    float ev = expf(sc[s] - mx);
    red[s] = ev; __syncthreads();
    for (int off = 128; off > 0; off >>= 1) { if (s < off) red[s] += red[s + off]; __syncthreads(); }
    float inv = 1.f / red[0]; __syncthreads();
    sc[s] = ev * inv; __syncthreads();
    if (s < HDIM) {
        float acc = 0.f;
        for (int t = 0; t < 256; ++t)
            acc += sc[t] * vh[((size_t)(b * 256 + t)) * 256 + h * HDIM + s];
        attn[(size_t)b * 256 + h * HDIM + s] = acc;
    }
}

__global__ void attn_out_k(const float* __restrict__ attn, const float* __restrict__ wo,
                           const float* __restrict__ bo, float* __restrict__ out)
{
    int n = threadIdx.x, b = blockIdx.x;
    float acc = bo[n];
    const float* a = attn + (size_t)b * 256;
    for (int d = 0; d < 256; ++d) acc += a[d] * wo[(size_t)d * 256 + n];
    out[(size_t)b * 256 + n] = acc;
}

__global__ void pre_build_k(const float* __restrict__ attno, const float* __restrict__ act,
                            h16* __restrict__ pre)
{
    int idx = blockIdx.x * 256 + threadIdx.x;            // b*1280+j
    int j = idx % 1280, b = idx / 1280;
    float v = (j < 256) ? attno[(size_t)b * 256 + j] : act[(size_t)b * DMODEL + (j - 256)];
    pre[idx] = (h16)v;
}

__global__ void glu_ln_k(const float* __restrict__ t, const float* __restrict__ g,
                         const float* __restrict__ bb, float* __restrict__ state)
{
    int b = blockIdx.x, tid = threadIdx.x;
    __shared__ float val[1024]; __shared__ float red[256];
    const float* tr = t + (size_t)b * 2048;
    float ls = 0.f;
    for (int j = tid; j < 1024; j += 256) {
        float a = tr[j], gg = tr[j + 1024];
        float v = a * (1.f / (1.f + expf(-gg)));
        val[j] = v; ls += v;
    }
    red[tid] = ls; __syncthreads();
    for (int off = 128; off > 0; off >>= 1) { if (tid < off) red[tid] += red[tid + off]; __syncthreads(); }
    float m = red[0] * (1.f / 1024.f); __syncthreads();
    float lq = 0.f;
    for (int j = tid; j < 1024; j += 256) { float d = val[j] - m; lq += d * d; }
    red[tid] = lq; __syncthreads();
    for (int off = 128; off > 0; off >>= 1) { if (tid < off) red[tid] += red[tid + off]; __syncthreads(); }
    float rstd = rsqrtf(red[0] * (1.f / 1024.f) + EPSV);
    for (int j = tid; j < 1024; j += 256)
        state[(size_t)b * 1024 + j] = g[j] * (val[j] - m) * rstd + bb[j];
}

__global__ void trace_shift_k(float* __restrict__ trace, const float* __restrict__ state)
{
    int idx = blockIdx.x * 256 + threadIdx.x;            // b*1024+d
    float* tr = trace + (size_t)idx * 25;
    for (int m = 0; m < 24; ++m) tr[m] = tr[m + 1];
    tr[24] = state[idx];
}

__global__ void tp1_k(const float* __restrict__ trace, const float* __restrict__ w,
                      const float* __restrict__ bb, float* __restrict__ hmid)
{
    size_t idx = (size_t)blockIdx.x * 256 + threadIdx.x; // b*1024*64 + d*64 + h
    int hh = (int)(idx & 63);
    size_t t2 = idx >> 6;
    int d = (int)(t2 & 1023);
    int b = (int)(t2 >> 10);
    const float* tr = trace + ((size_t)b * 1024 + d) * 25;
    float a = bb[(size_t)d * 128 + hh];
    float g = bb[(size_t)d * 128 + hh + 64];
#pragma unroll
    for (int m = 0; m < 25; ++m) {
        float tv = tr[m];
        a += tv * w[((size_t)m * 128 + hh) * 1024 + d];
        g += tv * w[((size_t)m * 128 + hh + 64) * 1024 + d];
    }
    hmid[idx] = a * (1.f / (1.f + expf(-g)));
}

__global__ void tp2_k(const float* __restrict__ hmid, const float* __restrict__ w,
                      const float* __restrict__ bb, float* __restrict__ act)
{
    int idx = blockIdx.x * 256 + threadIdx.x;            // b*1024+d
    int d = idx & 1023;
    const float* hm = hmid + (size_t)idx * 64;
    float a = bb[d * 2 + 0];
    float g = bb[d * 2 + 1];
#pragma unroll
    for (int h = 0; h < 64; ++h) {
        float v = hm[h];
        a += v * w[(size_t)(h * 2 + 0) * 1024 + d];
        g += v * w[(size_t)(h * 2 + 1) * 1024 + d];
    }
    act[idx] = a * (1.f / (1.f + expf(-g)));
}

__global__ void softent_k(const float* __restrict__ pred, float* __restrict__ predOut,
                          float* __restrict__ certOut, int iter)
{
    int b = blockIdx.x, tid = threadIdx.x;
    const float* p = pred + (size_t)b * OUTP;
    __shared__ float red[256];
    float mx = -1e30f;
    for (int n = tid; n < OUTD; n += 256) mx = fmaxf(mx, p[n]);
    red[tid] = mx; __syncthreads();
    for (int off = 128; off > 0; off >>= 1) { if (tid < off) red[tid] = fmaxf(red[tid], red[tid + off]); __syncthreads(); }
    mx = red[0]; __syncthreads();
    float s = 0.f;
    for (int n = tid; n < OUTD; n += 256) s += expf(p[n] - mx);
    red[tid] = s; __syncthreads();
    for (int off = 128; off > 0; off >>= 1) { if (tid < off) red[tid] += red[tid + off]; __syncthreads(); }
    float lse = mx + logf(red[0]); __syncthreads();
    float ent = 0.f;
    for (int n = tid; n < OUTD; n += 256) {
        float lp = p[n] - lse;
        ent += expf(lp) * lp;
        predOut[(size_t)b * OUTD * NITERS + (size_t)n * NITERS + iter] = p[n];
    }
    red[tid] = ent; __syncthreads();
    for (int off = 128; off > 0; off >>= 1) { if (tid < off) red[tid] += red[tid + off]; __syncthreads(); }
    if (tid == 0) {
        float ne = -red[0] / logf(1000.f);
        certOut[(size_t)b * 2 * NITERS + iter]          = ne;
        certOut[(size_t)b * 2 * NITERS + NITERS + iter] = 1.f - ne;
    }
}

// ---------------------------------------------------------------------------
extern "C" void kernel_launch(void* const* d_in, const int* in_sizes, int n_in,
                              void* d_out, int out_size, void* d_ws, size_t ws_size,
                              hipStream_t stream)
{
    (void)in_sizes; (void)n_in; (void)out_size; (void)ws_size;
    const float* x        = (const float*)d_in[0];
    const float* conv1_w  = (const float*)d_in[1];
    const float* conv1_b  = (const float*)d_in[2];
    const float* bn1_g    = (const float*)d_in[3];
    const float* bn1_b    = (const float*)d_in[4];
    const float* conv2_w  = (const float*)d_in[5];
    const float* conv2_b  = (const float*)d_in[6];
    const float* bn2_g    = (const float*)d_in[7];
    const float* bn2_b    = (const float*)d_in[8];
    const float* kv_w     = (const float*)d_in[9];
    const float* kv_b     = (const float*)d_in[10];
    const float* ln_kv_g  = (const float*)d_in[11];
    const float* ln_kv_b  = (const float*)d_in[12];
    const float* q_w      = (const float*)d_in[13];
    const float* q_b      = (const float*)d_in[14];
    const float* wq       = (const float*)d_in[15];
    const float* bq       = (const float*)d_in[16];
    const float* wk       = (const float*)d_in[17];
    const float* bk       = (const float*)d_in[18];
    const float* wv       = (const float*)d_in[19];
    const float* bv       = (const float*)d_in[20];
    const float* wo       = (const float*)d_in[21];
    const float* bo_p     = (const float*)d_in[22];
    const float* syn_w    = (const float*)d_in[23];
    const float* syn_b    = (const float*)d_in[24];
    const float* ln_syn_g = (const float*)d_in[25];
    const float* ln_syn_b = (const float*)d_in[26];
    const float* tp1_w    = (const float*)d_in[27];
    const float* tp1_b    = (const float*)d_in[28];
    const float* tp2_w    = (const float*)d_in[29];
    const float* tp2_b    = (const float*)d_in[30];
    const float* out_w    = (const float*)d_in[31];
    const float* out_b    = (const float*)d_in[32];
    const float* start_act   = (const float*)d_in[33];
    const float* start_trace = (const float*)d_in[34];
    const float* decay_a  = (const float*)d_in[35];
    const float* decay_o  = (const float*)d_in[36];

    float* outF    = (float*)d_out;
    float* predOut = outF;                        // (64,1000,16)
    float* certOut = outF + 1024000;              // (64,2,16)
    float* soOut   = outF + 1026048;              // (64,32896)

    char* ws = (char*)d_ws;
    size_t off = 0;
    auto alloc = [&](size_t bytes) -> void* {
        void* p = ws + off;
        off = (off + bytes + 255) & ~(size_t)255;
        return p;
    };

    float* f_y    = (float*)alloc((size_t)BATCH * 256 * 64 * 64 * 4);   // y1, reused for y2
    float* f_h1   = (float*)alloc((size_t)BATCH * 256 * 32 * 32 * 4);
    float* f_h2   = (float*)alloc((size_t)BATCH * 256 * 16 * 16 * 4);
    float* f_st   = (float*)alloc(512 * 4);
    h16*   h_feat = (h16*)  alloc((size_t)16384 * 256 * 2);
    h16*   h_wkv  = (h16*)  alloc((size_t)256 * 256 * 2);
    h16*   h_wk   = (h16*)  alloc((size_t)256 * 256 * 2);
    h16*   h_wv   = (h16*)  alloc((size_t)256 * 256 * 2);
    h16*   h_wq   = (h16*)  alloc((size_t)256 * 256 * 2);
    h16*   h_wsyn = (h16*)  alloc((size_t)2048 * 1280 * 2);
    h16*   h_wqw  = (h16*)  alloc((size_t)256 * SYNCH * 2);
    h16*   h_wout = (h16*)  alloc((size_t)OUTP * SYNCH * 2);
    float* f_kv   = (float*)alloc((size_t)16384 * 256 * 4);
    h16*   h_kv   = (h16*)  alloc((size_t)16384 * 256 * 2);
    float* f_kh   = (float*)alloc((size_t)16384 * 256 * 4);
    float* f_vh   = (float*)alloc((size_t)16384 * 256 * 4);
    int*   i_ti   = (int*)  alloc(SYNCH * 4);
    int*   i_tj   = (int*)  alloc(SYNCH * 4);
    float* f_ra   = (float*)alloc(SYNCH * 4);
    float* f_ro   = (float*)alloc(SYNCH * 4);
    float* f_aa   = (float*)alloc((size_t)BATCH * SYNCH * 4);
    float* f_ba   = (float*)alloc((size_t)BATCH * SYNCH * 4);
    float* f_ao   = (float*)alloc((size_t)BATCH * SYNCH * 4);
    float* f_bo   = (float*)alloc((size_t)BATCH * SYNCH * 4);
    h16*   h_sa   = (h16*)  alloc((size_t)BATCH * SYNCH * 2);
    h16*   h_so   = (h16*)  alloc((size_t)BATCH * SYNCH * 2);
    float* f_part = (float*)alloc((size_t)4194304 * 4);   // max split-K partials
    float* f_q    = (float*)alloc((size_t)BATCH * 256 * 4);
    h16*   h_q    = (h16*)  alloc((size_t)BATCH * 256 * 2);
    float* f_qh   = (float*)alloc((size_t)BATCH * 256 * 4);
    float* f_attn = (float*)alloc((size_t)BATCH * 256 * 4);
    float* f_atto = (float*)alloc((size_t)BATCH * 256 * 4);
    h16*   h_pre  = (h16*)  alloc((size_t)BATCH * 1280 * 2);
    float* f_t    = (float*)alloc((size_t)BATCH * 2048 * 4);
    float* f_state= (float*)alloc((size_t)BATCH * 1024 * 4);
    float* f_trace= (float*)alloc((size_t)BATCH * 1024 * 25 * 4);
    float* f_hmid = (float*)alloc((size_t)BATCH * 1024 * 64 * 4);
    float* f_act  = (float*)alloc((size_t)BATCH * 1024 * 4);
    float* f_pred = (float*)alloc((size_t)BATCH * OUTP * 4);

    auto gemm = [&](const h16* A, const h16* Bt, int M, int N, int K, int parts,
                    const float* bias, int biasN, float* oF, h16* oH) {
        int kSteps = K / 32;
        int spp = (kSteps + parts - 1) / parts;
        dim3 g(N / 256, M / 16, parts);
        gemm_wmma_splitk<<<g, dim3(128), 0, stream>>>(A, Bt, f_part, M, N, K, kSteps, spp);
        size_t MN = (size_t)M * N;
        int blocks = (int)((MN + 255) / 256);
        reduce_bias_k<<<dim3(blocks), dim3(256), 0, stream>>>(f_part, parts, MN, bias, biasN, oF, oH, N);
    };

    // ---- conv stem ----
    conv3x3_k<<<dim3(16, 256, BATCH), 256, 0, stream>>>(x, conv1_w, conv1_b, f_y, 3, 64);
    bn_stats_k<<<256, 256, 0, stream>>>(f_y, f_st, 4096);
    bn_relu_pool_k<<<65536, 256, 0, stream>>>(f_y, f_st, bn1_g, bn1_b, f_h1, 64);
    conv3x3_k<<<dim3(4, 256, BATCH), 256, 0, stream>>>(f_h1, conv2_w, conv2_b, f_y, 256, 32);
    bn_stats_k<<<256, 256, 0, stream>>>(f_y, f_st, 1024);
    bn_relu_pool_k<<<16384, 256, 0, stream>>>(f_y, f_st, bn2_g, bn2_b, f_h2, 32);
    feat16_k<<<16384, 256, 0, stream>>>(f_h2, h_feat);

    // ---- weight conversion (f16, [N][K], keeps big weights L2-resident) ----
    w_transpose_h<<<256,    256, 0, stream>>>(kv_w,  h_wkv,  256,   256,  256);
    w_transpose_h<<<256,    256, 0, stream>>>(wk,    h_wk,   256,   256,  256);
    w_transpose_h<<<256,    256, 0, stream>>>(wv,    h_wv,   256,   256,  256);
    w_transpose_h<<<256,    256, 0, stream>>>(wq,    h_wq,   256,   256,  256);
    w_transpose_h<<<10240,  256, 0, stream>>>(syn_w, h_wsyn, 1280,  2048, 2048);
    w_transpose_h<<<32896,  256, 0, stream>>>(q_w,   h_wqw,  SYNCH, 256,  256);
    w_transpose_h<<<131584, 256, 0, stream>>>(out_w, h_wout, SYNCH, OUTD, OUTP);

    // ---- KV path ----
    gemm(h_feat, h_wkv, 16384, 256, 256, 1, kv_b, 256, f_kv, nullptr);
    ln256_k<<<16384, 256, 0, stream>>>(f_kv, ln_kv_g, ln_kv_b, f_kv, h_kv);
    gemm(h_kv, h_wk, 16384, 256, 256, 1, bk, 256, f_kh, nullptr);
    gemm(h_kv, h_wv, 16384, 256, 256, 1, bv, 256, f_vh, nullptr);

    // ---- recurrent state init ----
    triu_init_k<<<129, 256, 0, stream>>>(i_ti, i_tj);
    exp_neg_k<<<129, 256, 0, stream>>>(decay_a, f_ra);
    exp_neg_k<<<129, 256, 0, stream>>>(decay_o, f_ro);
    init_state_k<<<256, 256, 0, stream>>>(start_act, start_trace, f_act, f_trace);
    init_syn_k<<<dim3(129, BATCH), 256, 0, stream>>>(start_act, i_ti, i_tj, f_aa, f_ba, f_ao, f_bo);

    // ---- 16 thought iterations ----
    for (int it = 0; it < NITERS; ++it) {
        synch_k<<<dim3(129, BATCH), 256, 0, stream>>>(f_act, 768, i_ti, i_tj, f_ra, f_aa, f_ba, nullptr, h_sa);
        gemm(h_sa, h_wqw, BATCH, 256, SYNCH, 64, q_b, 256, f_q, h_q);
        gemm(h_q, h_wq, BATCH, 256, 256, 1, bq, 256, f_qh, nullptr);
        attention_k<<<dim3(BATCH, 8), 256, 0, stream>>>(f_qh, f_kh, f_vh, f_attn);
        attn_out_k<<<BATCH, 256, 0, stream>>>(f_attn, wo, bo_p, f_atto);
        pre_build_k<<<320, 256, 0, stream>>>(f_atto, f_act, h_pre);
        gemm(h_pre, h_wsyn, BATCH, 2048, 1280, 1, syn_b, 2048, f_t, nullptr);
        glu_ln_k<<<BATCH, 256, 0, stream>>>(f_t, ln_syn_g, ln_syn_b, f_state);
        trace_shift_k<<<256, 256, 0, stream>>>(f_trace, f_state);
        tp1_k<<<16384, 256, 0, stream>>>(f_trace, tp1_w, tp1_b, f_hmid);
        tp2_k<<<256, 256, 0, stream>>>(f_hmid, tp2_w, tp2_b, f_act);
        synch_k<<<dim3(129, BATCH), 256, 0, stream>>>(f_act, 0, i_ti, i_tj, f_ro, f_ao, f_bo, soOut, h_so);
        gemm(h_so, h_wout, BATCH, OUTP, SYNCH, 64, out_b, OUTD, f_pred, nullptr);
        softent_k<<<BATCH, 256, 0, stream>>>(f_pred, predOut, certOut, it);
    }
}
